// TransformerLayer_29875792511774
// MI455X (gfx1250) — matrix operand order repaired and checked by reading, not compiled, not run on hardware
//
#include <hip/hip_runtime.h>
#include <hip/hip_bf16.h>
#include <math.h>

// ---------------------------------------------------------------------------
// Types for CDNA5 WMMA (wave32): bf16 16x16x32, f32 accumulate.
// ---------------------------------------------------------------------------
typedef __bf16 bf16;
typedef __attribute__((ext_vector_type(16))) __bf16 v16bf;
typedef __attribute__((ext_vector_type(8)))  __bf16 bf16x8;
typedef __attribute__((ext_vector_type(2)))  __bf16 bf16x2;
typedef __attribute__((ext_vector_type(8)))  float  v8f;

union FragU {
    v16bf  v;
    bf16x8 h[2];
};

// A-fragment (16x32 bf16). Per ISA: lane holds row = lane&15, half = lane>>4.
// element e <-> K = 16*half + e (this code's k order, shared with B) -> 16 contiguous bf16 at p.
__device__ __forceinline__ v16bf ld_frag_a(const bf16* p) {
    FragU u;
    u.h[0] = *(const bf16x8*)(p);
    u.h[1] = *(const bf16x8*)(p + 8);
    return u.v;
}

// B-fragment (32x16 bf16). Per ISA: lane holds col = lane&15,
// element e <-> k = 16*half + e (this code's k order, shared with A) -> 16 contiguous bf16 at p.
__device__ __forceinline__ v16bf ld_frag_b(const bf16* p) {
    FragU u;
    u.h[0] = *(const bf16x8*)(p);
    u.h[1] = *(const bf16x8*)(p + 8);
    return u.v;
}

__device__ __forceinline__ v8f wmma_bf16(v16bf a, v16bf b, v8f c) {
    return __builtin_amdgcn_wmma_f32_16x16x32_bf16(
        /*neg_a=*/false, a, /*neg_b=*/false, b,
        /*c_mod=*/(short)0, c, /*reuse_a=*/false, /*reuse_b=*/false);
}

// ---------------------------------------------------------------------------
// gfx1250 async copy: global -> LDS, 16 bytes per lane, tracked by ASYNCcnt.
// ---------------------------------------------------------------------------
__device__ __forceinline__ unsigned lds_off(const void* p) {
    // Flat address of an LDS object: low 32 bits are the LDS byte offset.
    return (unsigned)(unsigned long long)p;
}

__device__ __forceinline__ void async_copy_b128(unsigned lds, const void* gaddr) {
    asm volatile("global_load_async_to_lds_b128 %0, %1, off"
                 :: "v"(lds), "v"(gaddr) : "memory");
}

__device__ __forceinline__ void wait_async0() {
    asm volatile("s_wait_asynccnt 0" ::: "memory");
}

// ---------------------------------------------------------------------------
// fp32 -> bf16 conversion (weights)
// ---------------------------------------------------------------------------
__global__ void cvt_bf16_kernel(const float* __restrict__ in,
                                bf16* __restrict__ out, int n) {
    int i = blockIdx.x * blockDim.x + threadIdx.x;
    int stride = gridDim.x * blockDim.x;
    for (; i < n; i += stride) out[i] = (bf16)in[i];
}

// ---------------------------------------------------------------------------
// LayerNorm over D=512, one block (256 threads) per row; bf16 output.
// ---------------------------------------------------------------------------
__global__ __launch_bounds__(256) void layernorm_kernel(
    const float* __restrict__ x, const float* __restrict__ g,
    const float* __restrict__ b, bf16* __restrict__ out) {
    __shared__ float red[256];
    const int row = blockIdx.x;
    const int tid = threadIdx.x;
    const float* xr = x + (size_t)row * 512;

    float v0 = xr[tid];
    float v1 = xr[tid + 256];

    red[tid] = v0 + v1;
    __syncthreads();
    #pragma unroll
    for (int o = 128; o > 0; o >>= 1) {
        if (tid < o) red[tid] += red[tid + o];
        __syncthreads();
    }
    float mu = red[0] * (1.0f / 512.0f);
    __syncthreads();

    float d0 = v0 - mu, d1 = v1 - mu;
    red[tid] = d0 * d0 + d1 * d1;
    __syncthreads();
    #pragma unroll
    for (int o = 128; o > 0; o >>= 1) {
        if (tid < o) red[tid] += red[tid + o];
        __syncthreads();
    }
    float rs = rsqrtf(red[0] * (1.0f / 512.0f) + 1e-5f);

    bf16* outr = out + (size_t)row * 512;
    outr[tid]       = (bf16)(d0 * rs * g[tid] + b[tid]);
    outr[tid + 256] = (bf16)(d1 * rs * g[tid + 256] + b[tid + 256]);
}

// ---------------------------------------------------------------------------
// Generic bf16 WMMA GEMM: C[M,N] = A[M,K] * B[K,N] + bias (+ epilogue)
//   EPI = 0 : bias,            write bf16
//   EPI = 1 : bias + residual, write f32
//   EPI = 2 : bias + GELU,     write bf16
// Block tile 128x256, k-step 32. 256 threads = 8 waves in a 2x4 grid;
// each wave computes a 64x64 tile via 4x4 WMMA subtiles (16 WMMA / k-step).
// A tile staged with async-to-LDS; B tile staged transposed with packed
// ds_store_b32. M % 128 == 0, N % 256 == 0, K % 32 == 0 at all call sites.
// ---------------------------------------------------------------------------
template <int EPI>
__global__ __launch_bounds__(256) void gemm_bf16_kernel(
    const bf16* __restrict__ A, const bf16* __restrict__ B,
    const float* __restrict__ bias, const float* __restrict__ residual,
    float* __restrict__ outF, bf16* __restrict__ outB,
    int M, int N, int K) {
    __shared__ __align__(16) bf16 sA[128 * 32];   // [row][k]   8 KB
    __shared__ __align__(16) bf16 sBT[256 * 32];  // [col][k]  16 KB

    const int tid  = threadIdx.x;
    const int lane = tid & 31;
    const int wave = tid >> 5;
    const int half = lane >> 4;
    const int l16  = lane & 15;
    const int waveRow = wave >> 2;  // 0..1 -> 64-row slab
    const int waveCol = wave & 3;   // 0..3 -> 64-col slab
    const int blockRow = blockIdx.y * 128;
    const int blockCol = blockIdx.x * 256;

    v8f acc[4][4];
    #pragma unroll
    for (int sr = 0; sr < 4; ++sr)
        #pragma unroll
        for (int sc = 0; sc < 4; ++sc) {
            v8f z{};
            acc[sr][sc] = z;
        }

    for (int k0 = 0; k0 < K; k0 += 32) {
        // Stage A tile (128x32) via async global->LDS (2 x 16B per thread).
        {
            int row = tid >> 1;
            int seg = tid & 1;  // 16 k each
            const bf16* g = A + (size_t)(blockRow + row) * K + k0 + seg * 16;
            unsigned l = lds_off(sA + row * 32 + seg * 16);
            async_copy_b128(l, g);
            async_copy_b128(l + 16, g + 8);
        }
        // Stage B tile (32x256) transposed into [col][k] with packed b32
        // stores: each thread handles k = {2kp, 2kp+1} x 16 cols.
        {
            int kp  = tid & 15;
            int seg = tid >> 4;  // 0..15, 16 cols each
            const bf16* g0 = B + (size_t)(k0 + 2 * kp) * N + blockCol + seg * 16;
            const bf16* g1 = g0 + N;
            bf16x8 a0 = *(const bf16x8*)(g0);
            bf16x8 a1 = *(const bf16x8*)(g0 + 8);
            bf16x8 b0 = *(const bf16x8*)(g1);
            bf16x8 b1 = *(const bf16x8*)(g1 + 8);
            #pragma unroll
            for (int c = 0; c < 8; ++c) {
                bf16x2 pr; pr[0] = a0[c]; pr[1] = b0[c];
                *(bf16x2*)(sBT + (size_t)(seg * 16 + c) * 32 + 2 * kp) = pr;
            }
            #pragma unroll
            for (int c = 0; c < 8; ++c) {
                bf16x2 pr; pr[0] = a1[c]; pr[1] = b1[c];
                *(bf16x2*)(sBT + (size_t)(seg * 16 + 8 + c) * 32 + 2 * kp) = pr;
            }
        }
        wait_async0();
        __syncthreads();

        v16bf bfrag[4];
        #pragma unroll
        for (int sc = 0; sc < 4; ++sc)
            bfrag[sc] = ld_frag_b(sBT + (waveCol * 64 + sc * 16 + l16) * 32 + half * 16);

        #pragma unroll
        for (int sr = 0; sr < 4; ++sr) {
            v16bf afrag = ld_frag_a(sA + (waveRow * 64 + sr * 16 + l16) * 32 + half * 16);
            #pragma unroll
            for (int sc = 0; sc < 4; ++sc)
                acc[sr][sc] = wmma_bf16(afrag, bfrag[sc], acc[sr][sc]);
        }
        __syncthreads();
    }

    // Epilogue. D layout: element r of v8f -> row m = r + 8*half, col = lane&15.
    #pragma unroll
    for (int sr = 0; sr < 4; ++sr) {
        #pragma unroll
        for (int sc = 0; sc < 4; ++sc) {
            int col = blockCol + waveCol * 64 + sc * 16 + l16;
            float bc = bias[col];
            #pragma unroll
            for (int r = 0; r < 8; ++r) {
                int row = blockRow + waveRow * 64 + sr * 16 + half * 8 + r;
                size_t idx = (size_t)row * N + col;
                float v = acc[sr][sc][r] + bc;
                if (EPI == 1) {
                    outF[idx] = v + residual[idx];
                } else if (EPI == 2) {
                    v = 0.5f * v * (1.0f + erff(v * 0.70710678118654752f));
                    outB[idx] = (bf16)v;
                } else {
                    outB[idx] = (bf16)v;
                }
            }
        }
    }
}

// ---------------------------------------------------------------------------
// Causal flash attention. qkv is bf16 [4096][1536] (q|k|v, each H*64 cols).
// Grid: (S/64, H). 128 threads = 4 waves; wave w owns q rows qBase+16w..+15.
// Key blocks of 32; online softmax; output bf16 [4096][512].
// K tile staged via async-to-LDS; V tile transposed with packed b32 stores.
// ---------------------------------------------------------------------------
__global__ __launch_bounds__(128) void attn_kernel(
    const bf16* __restrict__ qkv, bf16* __restrict__ out) {
    __shared__ __align__(16) bf16 sK[32 * 64];      // [key][d]
    __shared__ __align__(16) bf16 sVT[64 * 32];     // [d][key]
    __shared__ __align__(16) bf16 sP[4][16 * 32];   // per-wave P tile [qr][key]

    const int tid  = threadIdx.x;
    const int lane = tid & 31;
    const int wave = tid >> 5;
    const int half = lane >> 4;
    const int l16  = lane & 15;
    const int qBase = blockIdx.x * 64;
    const int head  = blockIdx.y;
    const int qRow0 = qBase + wave * 16;

    // Load Q fragments (16x64 => two 16x32 A-frags), rows = qRow0 + (lane&15).
    // Fold in the softmax scale 1/sqrt(64) = 0.125 (exact in bf16).
    v16bf qfrag[2];
    {
        const bf16* qp = qkv + (size_t)(qRow0 + l16) * 1536 + head * 64;
        #pragma unroll
        for (int kk = 0; kk < 2; ++kk) {
            qfrag[kk] = ld_frag_a(qp + kk * 32 + half * 16);
            #pragma unroll
            for (int e = 0; e < 16; ++e)
                qfrag[kk][e] = (bf16)((float)qfrag[kk][e] * 0.125f);
        }
    }

    v8f o[4];
    float mstat[8], lstat[8];
    #pragma unroll
    for (int c = 0; c < 4; ++c) { v8f z{}; o[c] = z; }
    #pragma unroll
    for (int r = 0; r < 8; ++r) { mstat[r] = -3.0e38f; lstat[r] = 0.0f; }

    const int nj = blockIdx.x * 2 + 2;  // cover keys 0 .. qBase+63
    for (int j = 0; j < nj; ++j) {
        const int kb = j * 32;
        // Stage K [key][d] via async (2 x 16B per thread), V transposed with
        // packed b32 stores (keys 2kp,2kp+1 pack into one dword).
        {
            int key = tid >> 2;
            int seg = tid & 3;  // chunk ids seg and seg+4, 8 dims each
            const bf16* gk = qkv + (size_t)(kb + key) * 1536 + 512 + head * 64;
            unsigned lk = lds_off(sK + key * 64);
            async_copy_b128(lk + seg * 16, gk + seg * 8);
            async_copy_b128(lk + (seg + 4) * 16, gk + (seg + 4) * 8);

            int kp   = tid & 15;  // keys 2kp, 2kp+1
            int dseg = tid >> 4;  // 0..7, 8 dims each
            const bf16* gv0 = qkv + (size_t)(kb + 2 * kp) * 1536 + 1024 + head * 64 + dseg * 8;
            const bf16* gv1 = gv0 + 1536;
            bf16x8 d0 = *(const bf16x8*)(gv0);
            bf16x8 d1 = *(const bf16x8*)(gv1);
            #pragma unroll
            for (int c = 0; c < 8; ++c) {
                bf16x2 pr; pr[0] = d0[c]; pr[1] = d1[c];
                *(bf16x2*)(sVT + (size_t)(dseg * 8 + c) * 32 + 2 * kp) = pr;
            }
        }
        wait_async0();
        __syncthreads();

        // Scores: S(16x32) = Qs(16x64) @ K^T, two 16x16 N-tiles.
        v8f s0{}, s1{};
        #pragma unroll
        for (int kk = 0; kk < 2; ++kk) {
            v16bf b0 = ld_frag_b(sK + (0 * 16 + l16) * 64 + kk * 32 + half * 16);
            v16bf b1 = ld_frag_b(sK + (1 * 16 + l16) * 64 + kk * 32 + half * 16);
            s0 = wmma_bf16(qfrag[kk], b0, s0);
            s1 = wmma_bf16(qfrag[kk], b1, s1);
        }

        // Online softmax per row (row m = half*8 + r; col = lane&15).
        bf16* sp = sP[wave];
        #pragma unroll
        for (int r = 0; r < 8; ++r) {
            int qrow = qRow0 + half * 8 + r;
            float a0 = s0[r];
            float a1 = s1[r];
            if (kb + l16 > qrow)      a0 = -3.0e38f;
            if (kb + 16 + l16 > qrow) a1 = -3.0e38f;
            float mx = fmaxf(a0, a1);
            #pragma unroll
            for (int off = 8; off >= 1; off >>= 1)
                mx = fmaxf(mx, __shfl_xor(mx, off, 32));
            float mnew = fmaxf(mstat[r], mx);
            float p0 = __expf(a0 - mnew);
            float p1 = __expf(a1 - mnew);
            float rsum = p0 + p1;
            #pragma unroll
            for (int off = 8; off >= 1; off >>= 1)
                rsum += __shfl_xor(rsum, off, 32);
            float scl = __expf(mstat[r] - mnew);
            lstat[r] = lstat[r] * scl + rsum;
            mstat[r] = mnew;
            #pragma unroll
            for (int c = 0; c < 4; ++c) o[c][r] *= scl;
            int rr = half * 8 + r;
            sp[rr * 32 + l16]      = (bf16)p0;
            sp[rr * 32 + 16 + l16] = (bf16)p1;
        }
        __syncthreads();  // make per-wave P visible across lanes (D->A reshape)

        v16bf pfrag = ld_frag_a(sp + l16 * 32 + half * 16);
        #pragma unroll
        for (int c = 0; c < 4; ++c) {
            v16bf vfrag = ld_frag_b(sVT + (c * 16 + l16) * 32 + half * 16);
            o[c] = wmma_bf16(pfrag, vfrag, o[c]);
        }
        __syncthreads();  // before next iteration restages sK/sVT
    }

    // Normalize and write out (bf16 [row][512]).
    #pragma unroll
    for (int r = 0; r < 8; ++r) {
        int row = qRow0 + half * 8 + r;
        float inv = 1.0f / lstat[r];
        bf16* op = out + (size_t)row * 512 + head * 64 + l16;
        #pragma unroll
        for (int c = 0; c < 4; ++c)
            op[c * 16] = (bf16)(o[c][r] * inv);
    }
}

// ---------------------------------------------------------------------------
// Host-side orchestration.
// ---------------------------------------------------------------------------
static inline size_t align256(size_t x) { return (x + 255) & ~(size_t)255; }

extern "C" void kernel_launch(void* const* d_in, const int* in_sizes, int n_in,
                              void* d_out, int out_size, void* d_ws, size_t ws_size,
                              hipStream_t stream) {
    (void)in_sizes; (void)n_in; (void)out_size; (void)ws_size;
    const int S = 4096, D = 512, FF = 1024;

    const float* x     = (const float*)d_in[0];
    const float* w_qkv = (const float*)d_in[1];
    const float* b_qkv = (const float*)d_in[2];
    const float* w_out = (const float*)d_in[3];
    const float* b_out = (const float*)d_in[4];
    const float* w_ff1 = (const float*)d_in[5];
    const float* b_ff1 = (const float*)d_in[6];
    const float* w_ff2 = (const float*)d_in[7];
    const float* b_ff2 = (const float*)d_in[8];
    const float* g1    = (const float*)d_in[9];
    const float* bt1   = (const float*)d_in[10];
    const float* g2    = (const float*)d_in[11];
    const float* bt2   = (const float*)d_in[12];

    // Workspace layout (bf16 unless noted).
    char* ws = (char*)d_ws;
    size_t off = 0;
    bf16* h1      = (bf16*)(ws + off); off = align256(off + (size_t)S * D * 2);
    bf16* w_qkvb  = (bf16*)(ws + off); off = align256(off + (size_t)D * 3 * D * 2);
    bf16* qkvb    = (bf16*)(ws + off); off = align256(off + (size_t)S * 3 * D * 2);
    bf16* attnb   = (bf16*)(ws + off); off = align256(off + (size_t)S * D * 2);
    bf16* w_outb  = (bf16*)(ws + off); off = align256(off + (size_t)D * D * 2);
    float* x1     = (float*)(ws + off); off = align256(off + (size_t)S * D * 4);
    bf16* h2      = (bf16*)(ws + off); off = align256(off + (size_t)S * D * 2);
    bf16* w_ff1b  = (bf16*)(ws + off); off = align256(off + (size_t)D * FF * 2);
    bf16* ffb     = (bf16*)(ws + off); off = align256(off + (size_t)S * FF * 2);
    bf16* w_ff2b  = (bf16*)(ws + off); off = align256(off + (size_t)FF * D * 2);

    // Weight conversions.
    {
        int n;
        n = D * 3 * D; cvt_bf16_kernel<<<(n + 255) / 256, 256, 0, stream>>>(w_qkv, w_qkvb, n);
        n = D * D;     cvt_bf16_kernel<<<(n + 255) / 256, 256, 0, stream>>>(w_out, w_outb, n);
        n = D * FF;    cvt_bf16_kernel<<<(n + 255) / 256, 256, 0, stream>>>(w_ff1, w_ff1b, n);
        n = FF * D;    cvt_bf16_kernel<<<(n + 255) / 256, 256, 0, stream>>>(w_ff2, w_ff2b, n);
    }

    // LN1 -> h1 (bf16).
    layernorm_kernel<<<S, 256, 0, stream>>>(x, g1, bt1, h1);

    // QKV projection: [4096,512] x [512,1536] -> bf16 qkv.
    gemm_bf16_kernel<0><<<dim3(3 * D / 256, S / 128), 256, 0, stream>>>(
        h1, w_qkvb, b_qkv, nullptr, nullptr, qkvb, S, 3 * D, D);

    // Causal attention -> bf16 attn_out [4096,512].
    attn_kernel<<<dim3(S / 64, 8), 128, 0, stream>>>(qkvb, attnb);

    // Out projection + residual(x) -> f32 x1.
    gemm_bf16_kernel<1><<<dim3(D / 256, S / 128), 256, 0, stream>>>(
        attnb, w_outb, b_out, x, x1, nullptr, S, D, D);

    // LN2 -> h2 (bf16).
    layernorm_kernel<<<S, 256, 0, stream>>>(x1, g2, bt2, h2);

    // FF1 + GELU -> bf16 ffb [4096,1024].
    gemm_bf16_kernel<2><<<dim3(FF / 256, S / 128), 256, 0, stream>>>(
        h2, w_ff1b, b_ff1, nullptr, nullptr, ffb, S, FF, D);

    // FF2 + residual(x1) -> f32 d_out.
    gemm_bf16_kernel<1><<<dim3(D / 256, S / 128), 256, 0, stream>>>(
        ffb, w_ff2b, b_ff2, x1, (float*)d_out, nullptr, S, D, FF);
}
